// Decoder_5248450036465
// MI455X (gfx1250) — matrix operand order repaired and checked by reading, not compiled, not run on hardware
//
#include <hip/hip_runtime.h>
#include <hip/hip_bf16.h>
#include <math.h>

typedef __bf16 bf16_t;
typedef __attribute__((ext_vector_type(16))) __bf16 v16bf;
typedef __attribute__((ext_vector_type(8)))  float  v8f;
typedef __attribute__((ext_vector_type(4)))  unsigned int u32x4;

// 32 bytes per lane = one WMMA A/B bf16 fragment; loaded as 2x b128.
union Frag {
    v16bf v;
    struct { u32x4 lo, hi; } q;
};

__device__ __forceinline__ v8f wmma_bf16f32(v16bf a, v16bf b, v8f c) {
    // v_wmma_f32_16x16x32_bf16: D = A(16x32 bf16) x B(32x16 bf16) + C(16x16 f32)
    return __builtin_amdgcn_wmma_f32_16x16x32_bf16(false, a, false, b,
                                                   (short)0, c, false, false);
}

__device__ __forceinline__ float sigm(float x) { return 1.0f / (1.0f + __expf(-x)); }

// ---------------------------------------------------------------- conversions
__global__ void k_f32_to_bf16(const float* __restrict__ src,
                              bf16_t* __restrict__ dst, int n) {
    int i = blockIdx.x * blockDim.x + threadIdx.x;
    if (i < n) dst[i] = (bf16_t)src[i];
}

// Pack row-major f32 W[K,N] into per-lane WMMA B fragments:
// fragment (ntile, kt): lane L covers col = ntile*16 + (L&15),
// K = kt*32 + (L>>4)*16 + e  (e = 0..15, contiguous per lane).
__global__ void k_pack_b(const float* __restrict__ W, bf16_t* __restrict__ out,
                         int K, int N) {
    int gid  = blockIdx.x * blockDim.x + threadIdx.x;
    int lane = gid & 31;
    int frag = gid >> 5;
    int kfr  = K >> 5;
    int nfrags = (N >> 4) * kfr;
    if (frag >= nfrags) return;
    int ntile = frag / kfr;
    int kt    = frag - ntile * kfr;
    int col   = (ntile << 4) + (lane & 15);
    int kbase = (kt << 5) + ((lane >> 4) << 4);
    bf16_t* dst = out + (size_t)frag * 512 + lane * 16;
#pragma unroll
    for (int e = 0; e < 16; ++e)
        dst[e] = (bf16_t)W[(size_t)(kbase + e) * N + col];
}

__global__ void k_init_state(const float* __restrict__ c0,
                             const float* __restrict__ h0,
                             float* __restrict__ c_state,
                             bf16_t* __restrict__ hbuf, int n) {
    int i = blockIdx.x * blockDim.x + threadIdx.x;
    if (i < n) { c_state[i] = c0[i]; hbuf[i] = (bf16_t)h0[i]; }
}

// --------------------------------------------------------------- bulk GEMM
// C[M,N] = A[M,K](bf16 row-major) x Bp(packed bf16 fragments) + bias[N], f32 out.
// One wave -> 16(M) x 64(N) strip. Software-pipelined: iteration k+1's
// fragment loads are issued before the WMMAs consuming iteration k.
__global__ void k_gemm_bf16(const bf16_t* __restrict__ A,
                            const bf16_t* __restrict__ Bp,
                            const float* __restrict__ bias,
                            float* __restrict__ C, int N, int K) {
    int w    = (blockIdx.x * blockDim.x + threadIdx.x) >> 5;
    int lane = threadIdx.x & 31;
    int nstrips = N >> 6;
    int mt = w / nstrips;
    int ns = w - mt * nstrips;
    int kfr = K >> 5;

    v8f accs[4] = {};
    int arow = (mt << 4) + (lane & 15);
    int kb   = (lane >> 4) << 4;
    const bf16_t* ap = A + (size_t)arow * K + kb;
    size_t fb     = ((size_t)(ns << 2) * kfr) * 512 + (size_t)lane * 16;
    size_t stride = (size_t)kfr * 512;

    // prologue: load k-slice 0
    Frag a, b0, b1, b2, b3;
    a.q.lo = *(const u32x4*)(ap);
    a.q.hi = *(const u32x4*)(ap + 8);
    {
        const bf16_t* p = Bp + fb;
        b0.q.lo = *(const u32x4*)(p);              b0.q.hi = *(const u32x4*)(p + 8);
        b1.q.lo = *(const u32x4*)(p + stride);     b1.q.hi = *(const u32x4*)(p + stride + 8);
        b2.q.lo = *(const u32x4*)(p + 2*stride);   b2.q.hi = *(const u32x4*)(p + 2*stride + 8);
        b3.q.lo = *(const u32x4*)(p + 3*stride);   b3.q.hi = *(const u32x4*)(p + 3*stride + 8);
    }

    for (int kt = 0; kt < kfr - 1; ++kt) {
        ap += 32; fb += 512;
        Frag an, bn0, bn1, bn2, bn3;
        an.q.lo = *(const u32x4*)(ap);
        an.q.hi = *(const u32x4*)(ap + 8);
        const bf16_t* p = Bp + fb;
        bn0.q.lo = *(const u32x4*)(p);              bn0.q.hi = *(const u32x4*)(p + 8);
        bn1.q.lo = *(const u32x4*)(p + stride);     bn1.q.hi = *(const u32x4*)(p + stride + 8);
        bn2.q.lo = *(const u32x4*)(p + 2*stride);   bn2.q.hi = *(const u32x4*)(p + 2*stride + 8);
        bn3.q.lo = *(const u32x4*)(p + 3*stride);   bn3.q.hi = *(const u32x4*)(p + 3*stride + 8);

        accs[0] = wmma_bf16f32(a.v, b0.v, accs[0]);
        accs[1] = wmma_bf16f32(a.v, b1.v, accs[1]);
        accs[2] = wmma_bf16f32(a.v, b2.v, accs[2]);
        accs[3] = wmma_bf16f32(a.v, b3.v, accs[3]);

        a = an; b0 = bn0; b1 = bn1; b2 = bn2; b3 = bn3;
    }
    accs[0] = wmma_bf16f32(a.v, b0.v, accs[0]);
    accs[1] = wmma_bf16f32(a.v, b1.v, accs[1]);
    accs[2] = wmma_bf16f32(a.v, b2.v, accs[2]);
    accs[3] = wmma_bf16f32(a.v, b3.v, accs[3]);

    int rowhi = (lane >> 4) << 3;
    int colb  = lane & 15;
#pragma unroll
    for (int j = 0; j < 4; ++j) {
        int col = (ns << 6) + (j << 4) + colb;
        float bv = bias[col];
#pragma unroll
        for (int r = 0; r < 8; ++r) {
            int row = (mt << 4) + rowhi + r;
            C[(size_t)row * N + col] = accs[j][r] + bv;
        }
    }
}

// ----------------------------------------------------------- LSTM time step
// grid = 64 blocks (16 H-columns each), block = 128 threads = 4 waves (one gate each).
// Latency-bound (only 256 waves chip-wide) -> software-pipelined fragment loads.
__global__ void k_lstm_step(const bf16_t* __restrict__ h_prev,
                            const bf16_t* __restrict__ WhP,
                            const float* __restrict__ Z,
                            float* __restrict__ c_state,
                            bf16_t* __restrict__ h_next,
                            bf16_t* __restrict__ hs,
                            float* __restrict__ out_c,
                            float* __restrict__ out_h,
                            int t, int is_last) {
    __shared__ float lds[4][32][16];
    int tid  = threadIdx.x;
    int lane = tid & 31;
    int g    = tid >> 5;        // gate: 0=i 1=f 2=g 3=o
    int nt   = blockIdx.x;      // H column tile (16 cols)

    v8f acc0 = {}, acc1 = {};
    int r  = lane & 15;
    int kb = (lane >> 4) << 4;
    const bf16_t* a0 = h_prev + (size_t)r * 1024 + kb;          // rows 0..15
    const bf16_t* a1 = h_prev + (size_t)(16 + r) * 1024 + kb;   // rows 16..31
    int ntg = (g << 6) + nt;                                    // tile in 4H space
    const bf16_t* bp = WhP + (size_t)ntg * 32 * 512 + (size_t)lane * 16;

    Frag fa0, fa1, fb;
    fa0.q.lo = *(const u32x4*)(a0);  fa0.q.hi = *(const u32x4*)(a0 + 8);
    fa1.q.lo = *(const u32x4*)(a1);  fa1.q.hi = *(const u32x4*)(a1 + 8);
    fb.q.lo  = *(const u32x4*)(bp);  fb.q.hi  = *(const u32x4*)(bp + 8);

#pragma unroll 4
    for (int kt = 0; kt < 31; ++kt) {
        a0 += 32; a1 += 32; bp += 512;
        Frag na0, na1, nb;
        na0.q.lo = *(const u32x4*)(a0);  na0.q.hi = *(const u32x4*)(a0 + 8);
        na1.q.lo = *(const u32x4*)(a1);  na1.q.hi = *(const u32x4*)(a1 + 8);
        nb.q.lo  = *(const u32x4*)(bp);  nb.q.hi  = *(const u32x4*)(bp + 8);

        acc0 = wmma_bf16f32(fa0.v, fb.v, acc0);
        acc1 = wmma_bf16f32(fa1.v, fb.v, acc1);

        fa0 = na0; fa1 = na1; fb = nb;
    }
    acc0 = wmma_bf16f32(fa0.v, fb.v, acc0);
    acc1 = wmma_bf16f32(fa1.v, fb.v, acc1);

    int rowhi = (lane >> 4) << 3;
    int colb  = lane & 15;
#pragma unroll
    for (int rr = 0; rr < 8; ++rr) {
        lds[g][rowhi + rr][colb]      = acc0[rr];
        lds[g][16 + rowhi + rr][colb] = acc1[rr];
    }
    __syncthreads();

    for (int e = tid; e < 512; e += 128) {
        int b  = e >> 4, jc = e & 15;
        int colh = (nt << 4) + jc;
        size_t zrow = ((size_t)b * 256 + t) * 4096;
        float zi = Z[zrow + colh]        + lds[0][b][jc];
        float zf = Z[zrow + 1024 + colh] + lds[1][b][jc];
        float zg = Z[zrow + 2048 + colh] + lds[2][b][jc];
        float zo = Z[zrow + 3072 + colh] + lds[3][b][jc];
        float iv = sigm(zi), fv = sigm(zf), gv = tanhf(zg), ov = sigm(zo);
        int cidx = (b << 10) + colh;
        float cn = fv * c_state[cidx] + iv * gv;
        c_state[cidx] = cn;
        float hn = ov * tanhf(cn);
        bf16_t hb = (bf16_t)hn;
        h_next[cidx] = hb;
        hs[((size_t)b * 256 + t) * 1024 + colh] = hb;
        if (is_last) { out_c[cidx] = cn; out_h[cidx] = hn; }
    }
}

// ------------------------------------------------------------------- launch
extern "C" void kernel_launch(void* const* d_in, const int* in_sizes, int n_in,
                              void* d_out, int out_size, void* d_ws, size_t ws_size,
                              hipStream_t stream) {
    const float* c0     = (const float*)d_in[0];
    const float* h0     = (const float*)d_in[1];
    const float* inputs = (const float*)d_in[2];
    const float* Wi     = (const float*)d_in[3];
    const float* Wh     = (const float*)d_in[4];
    const float* b      = (const float*)d_in[5];
    const float* Wd     = (const float*)d_in[6];
    const float* bd     = (const float*)d_in[7];

    const int B = 32, T = 256, V = 1024, H = 1024;
    const int BT = B * T; // 8192

    char* ws = (char*)d_ws;
    size_t off = 0;
    bf16_t* Xb  = (bf16_t*)(ws + off); off += (size_t)BT * V * 2;      // 16 MB
    bf16_t* WiP = (bf16_t*)(ws + off); off += (size_t)V * 4 * H * 2;   //  8 MB
    bf16_t* WhP = (bf16_t*)(ws + off); off += (size_t)H * 4 * H * 2;   //  8 MB
    bf16_t* WdP = (bf16_t*)(ws + off); off += (size_t)H * V * 2;       //  2 MB
    float*  Zb  = (float*) (ws + off); off += (size_t)BT * 4 * H * 4;  // 128 MB
    bf16_t* HS  = (bf16_t*)(ws + off); off += (size_t)BT * H * 2;      // 16 MB
    float*  Cst = (float*) (ws + off); off += (size_t)B * H * 4;
    bf16_t* Hb0 = (bf16_t*)(ws + off); off += (size_t)B * H * 2;
    bf16_t* Hb1 = (bf16_t*)(ws + off); off += (size_t)B * H * 2;

    // 1. inputs -> bf16
    {
        int n = BT * V;
        k_f32_to_bf16<<<(n + 255) / 256, 256, 0, stream>>>(inputs, Xb, n);
    }
    // 2. pack weights into WMMA B-fragment layout
    {
        int thr = (4 * H / 16) * (V / 32) * 32;
        k_pack_b<<<(thr + 255) / 256, 256, 0, stream>>>(Wi, WiP, V, 4 * H);
        thr = (4 * H / 16) * (H / 32) * 32;
        k_pack_b<<<(thr + 255) / 256, 256, 0, stream>>>(Wh, WhP, H, 4 * H);
        thr = (V / 16) * (H / 32) * 32;
        k_pack_b<<<(thr + 255) / 256, 256, 0, stream>>>(Wd, WdP, H, V);
    }
    // 3. init c/h state
    k_init_state<<<(B * H + 255) / 256, 256, 0, stream>>>(c0, h0, Cst, Hb0, B * H);
    // 4. Z = inputs @ Wi + b   (M=8192, N=4096, K=1024)
    {
        int waves = (BT / 16) * (4 * H / 64);
        k_gemm_bf16<<<waves * 32 / 256, 256, 0, stream>>>(Xb, WiP, b, Zb, 4 * H, V);
    }
    // 5. sequential scan (kernel launch per step = global sync; graph replays it)
    float* out_c = (float*)d_out;
    float* out_h = out_c + B * H;
    float* out_x = out_h + B * H;
    for (int t = 0; t < T; ++t) {
        const bf16_t* hin  = (t & 1) ? Hb1 : Hb0;
        bf16_t*       hout = (t & 1) ? Hb0 : Hb1;
        k_lstm_step<<<H / 16, 128, 0, stream>>>(hin, WhP, Zb, Cst, hout, HS,
                                                out_c, out_h, t, (t == T - 1) ? 1 : 0);
    }
    // 6. x = hs @ Wd + bd   (M=8192, N=1024, K=1024)
    {
        int waves = (BT / 16) * (V / 64);
        k_gemm_bf16<<<waves * 32 / 256, 256, 0, stream>>>(HS, WdP, bd, out_x, V, H);
    }
}
